// IntegratedChromaTransformer2DModel_49185965474069
// MI455X (gfx1250) — matrix operand order repaired and checked by reading, not compiled, not run on hardware
//
#include <hip/hip_runtime.h>
#include <hip/hip_bf16.h>
#include <math.h>

typedef _Float16 f16;
typedef __attribute__((ext_vector_type(16))) _Float16 v16h;
typedef __attribute__((ext_vector_type(8)))  _Float16 v8h;
typedef __attribute__((ext_vector_type(8)))  float    v8f;

#define HID   3072
#define MLPD  12288
#define NH    24
#define HDIM  128
#define LI    1024
#define LT    256
#define LL    1280
#define APXD  5120
#define MODL  17

__device__ __forceinline__ float gelu_f(float x) {
  return 0.5f * x * (1.0f + tanhf(0.7978845608028654f * (x + 0.044715f * x * x * x)));
}
__device__ __forceinline__ float silu_f(float x) {
  return x / (1.0f + __expf(-x));
}

// ---------------------------------------------------------------------------
// Weight convert + transpose: src f32 [K][N] -> dst f16 [N][K]
// ---------------------------------------------------------------------------
__global__ __launch_bounds__(256)
void wt_kernel(const float* __restrict__ src, f16* __restrict__ dst, int K, int N)
{
  __shared__ float tile[32][33];
  const int tk = blockIdx.y * 32, tn = blockIdx.x * 32;
  const int r = threadIdx.x >> 5, c = threadIdx.x & 31;
#pragma unroll
  for (int i = 0; i < 32; i += 8) {
    int k = tk + r + i, n = tn + c;
    tile[r + i][c] = (k < K && n < N) ? src[(size_t)k * N + n] : 0.0f;
  }
  __syncthreads();
#pragma unroll
  for (int i = 0; i < 32; i += 8) {
    int n = tn + r + i, k = tk + c;
    if (n < N && k < K) dst[(size_t)n * K + k] = (f16)tile[c][r + i];
  }
}

// ---------------------------------------------------------------------------
// WMMA GEMM: C[M][N](+epilogue) = A_f16[M][K] * Bt_f16[N][K]
// Block = 256 thr (8 waves), tile 128x128, K-step 32.
// Double-buffered LDS A tile; B fragments register-pipelined from global.
// Out-of-range rows/cols handled by pointer clamping (contaminated WMMA
// outputs land only in rows/cols that the epilogue never stores).
// ---------------------------------------------------------------------------
__global__ __launch_bounds__(256)
void gemm_f16_kernel(const f16* __restrict__ A, const f16* __restrict__ Bt,
                     void* __restrict__ Cout,
                     const float* __restrict__ bias, const float* __restrict__ gate,
                     const float* __restrict__ res,
                     int M, int N, int K, int ldc,
                     long sA, long sB, long sC,
                     int act, int outHalf, int doClip)
{
  __shared__ f16 As[2][128 * 40];
  const int z = blockIdx.z;
  A  += (size_t)z * sA;
  Bt += (size_t)z * sB;
  const size_t cOff = (size_t)z * sC;
  const int mBase = blockIdx.y * 128;
  const int nBase = blockIdx.x * 128;
  const int tid   = threadIdx.x;
  const int wave  = tid >> 5, lane = tid & 31;
  const int lhalf = lane >> 4, l16 = lane & 15;

  v8f acc[8] = {};

  // B: this wave's 16 output columns, clamped into range.
  const int  n    = nBase + wave * 16 + l16;
  const bool nOK  = (n < N);
  const f16* bBase = Bt + (size_t)(nOK ? n : (N - 1)) * K + lhalf * 16;

  // A staging: 2 threads per row, 16 halves each.
  const int  arow = tid >> 1;
  const int  acol = (tid & 1) * 16;
  const int  gm   = mBase + arow;
  const f16* aSrc = A + (size_t)(gm < M ? gm : (M - 1)) * K + acol;

  const int nsteps = K >> 5;

  union BU { v16h v; v8h h[2]; };

  // prologue: stage tile 0, load B fragment 0
  {
    v8h x0 = *(const v8h*)(aSrc);
    v8h x1 = *(const v8h*)(aSrc + 8);
    *(v8h*)&As[0][arow * 40 + acol]     = x0;
    *(v8h*)&As[0][arow * 40 + acol + 8] = x1;
  }
  BU bu, bnx;
  bu.h[0] = *(const v8h*)(bBase);
  bu.h[1] = *(const v8h*)(bBase + 8);
  bnx = bu;
  __syncthreads();

  for (int i = 0; i < nsteps; ++i) {
    const int  cur  = i & 1;
    const f16* abuf = As[cur];
    if (i + 1 < nsteps) {
      const int kb2 = (i + 1) << 5;
      v8h x0 = *(const v8h*)(aSrc + kb2);
      v8h x1 = *(const v8h*)(aSrc + kb2 + 8);
      *(v8h*)&As[cur ^ 1][arow * 40 + acol]     = x0;
      *(v8h*)&As[cur ^ 1][arow * 40 + acol + 8] = x1;
      bnx.h[0] = *(const v8h*)(bBase + kb2);
      bnx.h[1] = *(const v8h*)(bBase + kb2 + 8);
      __builtin_prefetch((const void*)(bBase + kb2 + 32), 0, 1);
    }
    BU au[2];
    {
      const f16* ap0 = abuf + l16 * 40;
      au[0].h[0] = *(const v8h*)(ap0 + lhalf * 16);
      au[0].h[1] = *(const v8h*)(ap0 + lhalf * 16 + 8);
    }
#pragma unroll
    for (int ms = 0; ms < 8; ++ms) {
      if (ms + 1 < 8) {
        const f16* ap = abuf + ((ms + 1) * 16 + l16) * 40;
        au[(ms + 1) & 1].h[0] = *(const v8h*)(ap + lhalf * 16);
        au[(ms + 1) & 1].h[1] = *(const v8h*)(ap + lhalf * 16 + 8);
      }
      acc[ms] = __builtin_amdgcn_wmma_f32_16x16x32_f16(
          false, au[ms & 1].v, false, bu.v, (short)0, acc[ms], false, false);
    }
    bu = bnx;
    __syncthreads();
  }

  if (!nOK) return;
  const float bv = bias ? bias[n] : 0.0f;
  const float gv = gate ? gate[n] : 1.0f;
#pragma unroll
  for (int ms = 0; ms < 8; ++ms) {
#pragma unroll
    for (int r = 0; r < 8; ++r) {
      const int row = mBase + ms * 16 + lhalf * 8 + r;
      if (row >= M) continue;
      float v = acc[ms][r] + bv;
      if (act == 1)      v = gelu_f(v);
      else if (act == 2) v = silu_f(v);
      const size_t idx = cOff + (size_t)row * ldc + n;
      if (res)    v = res[idx] + gv * v;
      if (doClip) v = fminf(fmaxf(v, -65504.0f), 65504.0f);
      if (outHalf) ((f16*)Cout)[idx] = (f16)v;
      else         ((float*)Cout)[idx] = v;
    }
  }
}

// ---------------------------------------------------------------------------
// LayerNorm (mode 0: shift + (1+scale)*ln(x)) or RMS (mode 1: rms(x)*scale)
// One block (256 thr) per row; writes f16.
// ---------------------------------------------------------------------------
__global__ __launch_bounds__(256)
void norm_mod_kernel(const float* __restrict__ src, f16* __restrict__ dst,
                     const float* __restrict__ shift, const float* __restrict__ scale,
                     int D, int mode)
{
  const int row = blockIdx.x;
  src += (size_t)row * D; dst += (size_t)row * D;
  __shared__ float red[256];
  const int tid = threadIdx.x;
  float mu = 0.0f;
  if (mode == 0) {
    float s = 0.0f;
    for (int i = tid; i < D; i += 256) s += src[i];
    red[tid] = s; __syncthreads();
    for (int st = 128; st > 0; st >>= 1) { if (tid < st) red[tid] += red[tid + st]; __syncthreads(); }
    mu = red[0] / (float)D;
    __syncthreads();
  }
  float v = 0.0f;
  for (int i = tid; i < D; i += 256) { float x = src[i] - mu; v += x * x; }
  red[tid] = v; __syncthreads();
  for (int st = 128; st > 0; st >>= 1) { if (tid < st) red[tid] += red[tid + st]; __syncthreads(); }
  const float rs = rsqrtf(red[0] / (float)D + 1e-6f);
  for (int i = tid; i < D; i += 256) {
    float x = (src[i] - mu) * rs;
    float o = (mode == 0) ? (shift[i] + (1.0f + scale[i]) * x) : (x * scale[i]);
    dst[i] = (f16)o;
  }
}

// ---------------------------------------------------------------------------
// QKV post: per (token, head) rms-norm q/k, RoPE, write Q/K f16 [H][L][128],
// V transposed f16 [H][128][L].  grid (LL, NH), block 128.
// ---------------------------------------------------------------------------
__global__ __launch_bounds__(128)
void qkv_post_kernel(const float* __restrict__ QKV, int ld,
                     const float* __restrict__ img_ids,
                     const float* __restrict__ qs_t, const float* __restrict__ ks_t,
                     const float* __restrict__ qs_i, const float* __restrict__ ks_i,
                     f16* __restrict__ Q, f16* __restrict__ Kd, f16* __restrict__ Vt)
{
  const int t = blockIdx.x;
  const int h = blockIdx.y;
  const int d = threadIdx.x;
  const float* base = QKV + (size_t)t * ld;
  float q = base[h * HDIM + d];
  float k = base[HID + h * HDIM + d];
  float v = base[2 * HID + h * HDIM + d];
  const float* qs = (t < LT) ? qs_t : qs_i;
  const float* ks = (t < LT) ? ks_t : ks_i;

  __shared__ float red[HDIM];
  red[d] = q * q; __syncthreads();
  for (int s = 64; s > 0; s >>= 1) { if (d < s) red[d] += red[d + s]; __syncthreads(); }
  q *= rsqrtf(red[0] * (1.0f / HDIM) + 1e-6f) * qs[d];
  __syncthreads();
  red[d] = k * k; __syncthreads();
  for (int s = 64; s > 0; s >>= 1) { if (d < s) red[d] += red[d + s]; __syncthreads(); }
  k *= rsqrtf(red[0] * (1.0f / HDIM) + 1e-6f) * ks[d];
  __syncthreads();

  __shared__ float qsh[HDIM], ksh[HDIM];
  qsh[d] = q; ksh[d] = k;
  __syncthreads();
  if (d < 64) {
    int a, j, dim;
    if (d < 8)       { a = 0; j = d;      dim = 16; }
    else if (d < 36) { a = 1; j = d - 8;  dim = 56; }
    else             { a = 2; j = d - 36; dim = 56; }
    float pos = 0.0f;
    if (t >= LT) pos = img_ids[(size_t)(t - LT) * 3 + a];
    const float om  = powf(10000.0f, -2.0f * (float)j / (float)dim);
    const float ang = pos * om;
    const float c = cosf(ang), s = sinf(ang);
    const size_t o = ((size_t)h * LL + t) * HDIM;
    float x0 = qsh[2 * d], x1 = qsh[2 * d + 1];
    Q[o + 2 * d]     = (f16)(c * x0 - s * x1);
    Q[o + 2 * d + 1] = (f16)(s * x0 + c * x1);
    x0 = ksh[2 * d]; x1 = ksh[2 * d + 1];
    Kd[o + 2 * d]     = (f16)(c * x0 - s * x1);
    Kd[o + 2 * d + 1] = (f16)(s * x0 + c * x1);
  }
  Vt[((size_t)h * HDIM + d) * LL + t] = (f16)v;
}

// ---------------------------------------------------------------------------
// Row softmax over L=1280 keys, batched per head.  grid (LL, NH), block 256.
// ---------------------------------------------------------------------------
__global__ __launch_bounds__(256)
void softmax_kernel(const float* __restrict__ S, f16* __restrict__ P, float scale)
{
  const int row = blockIdx.x, z = blockIdx.y;
  const float* s = S + ((size_t)z * LL + row) * LL;
  f16* p = P + ((size_t)z * LL + row) * LL;
  __shared__ float red[256];
  const int tid = threadIdx.x;
  float mx = -3.0e38f;
  for (int i = tid; i < LL; i += 256) mx = fmaxf(mx, s[i] * scale);
  red[tid] = mx; __syncthreads();
  for (int st = 128; st > 0; st >>= 1) { if (tid < st) red[tid] = fmaxf(red[tid], red[tid + st]); __syncthreads(); }
  mx = red[0]; __syncthreads();
  float sum = 0.0f;
  for (int i = tid; i < LL; i += 256) sum += __expf(s[i] * scale - mx);
  red[tid] = sum; __syncthreads();
  for (int st = 128; st > 0; st >>= 1) { if (tid < st) red[tid] += red[tid + st]; __syncthreads(); }
  const float inv = 1.0f / red[0];
  for (int i = tid; i < LL; i += 256) p[i] = (f16)(__expf(s[i] * scale - mx) * inv);
}

// ---------------------------------------------------------------------------
// Elementwise f32 -> f16 with optional gelu, strided src/dst.
// ---------------------------------------------------------------------------
__global__ __launch_bounds__(256)
void ew_cvt_kernel(const float* __restrict__ src, long sld,
                   f16* __restrict__ dst, long dld,
                   int cols, long total, int act)
{
  long i = (long)blockIdx.x * 256 + threadIdx.x;
  if (i >= total) return;
  long r = i / cols; int c = (int)(i % cols);
  float v = src[r * sld + c];
  if (act == 1) v = gelu_f(v);
  dst[r * dld + c] = (f16)v;
}

__global__ __launch_bounds__(256)
void clip_kernel(float* __restrict__ x, long n)
{
  long i = (long)blockIdx.x * 256 + threadIdx.x;
  if (i < n) x[i] = fminf(fmaxf(x[i], -65504.0f), 65504.0f);
}

// ---------------------------------------------------------------------------
// Approximator input embedding: out f32 [17][64] = [t_emb(ts,16), t_emb(gd,16),
// t_emb(row,32)]
// ---------------------------------------------------------------------------
__global__ __launch_bounds__(256)
void emb_kernel(const float* __restrict__ ts, const float* __restrict__ gd,
                float* __restrict__ out)
{
  int i = blockIdx.x * 256 + threadIdx.x;
  if (i >= MODL * 64) return;
  int j = i / 64, c = i % 64;
  float val;
  if (c < 32) {
    float t = (c < 16 ? ts[0] : gd[0]) * 1000.0f;
    int cc = c & 15;
    float fr = powf(10000.0f, -(float)(cc & 7) / 8.0f);
    float a = t * fr;
    val = (cc < 8) ? cosf(a) : sinf(a);
  } else {
    float t = (float)j * 1000.0f;
    int cc = c - 32;
    float fr = powf(10000.0f, -(float)(cc & 15) / 16.0f);
    float a = t * fr;
    val = (cc < 16) ? cosf(a) : sinf(a);
  }
  out[i] = val;
}

// ---------------------------------------------------------------------------
// Host orchestration
// ---------------------------------------------------------------------------
extern "C" void kernel_launch(void* const* d_in, const int* in_sizes, int n_in,
                              void* d_out, int out_size, void* d_ws, size_t ws_size,
                              hipStream_t stream)
{
  (void)in_sizes; (void)n_in; (void)out_size; (void)ws_size;
#define PARAM(i) ((const float*)d_in[i])
  const float* in_img     = PARAM(0);
  const float* in_img_ids = PARAM(1);
  const float* in_txt     = PARAM(2);
  const float* in_ts      = PARAM(4);
  const float* in_gd      = PARAM(5);
  // params (dict order)
  const float* img_in_w   = PARAM(6);  const float* img_in_b  = PARAM(7);
  const float* txt_in_w   = PARAM(8);  const float* txt_in_b  = PARAM(9);
  const float* apx_in_w   = PARAM(10); const float* apx_in_b  = PARAM(11);
  const float* apx_norm_s = PARAM(12);
  const float* apx_l_in_w = PARAM(13); const float* apx_l_in_b  = PARAM(14);
  const float* apx_l_out_w= PARAM(15); const float* apx_l_out_b = PARAM(16);
  const float* apx_out_w  = PARAM(17); const float* apx_out_b   = PARAM(18);
  const float* di_qkv_w   = PARAM(19); const float* di_qkv_b  = PARAM(20);
  const float* di_q_s     = PARAM(21); const float* di_k_s    = PARAM(22);
  const float* di_proj_w  = PARAM(23); const float* di_proj_b = PARAM(24);
  const float* di_mlp0_w  = PARAM(25); const float* di_mlp0_b = PARAM(26);
  const float* di_mlp2_w  = PARAM(27); const float* di_mlp2_b = PARAM(28);
  const float* dt_qkv_w   = PARAM(29); const float* dt_qkv_b  = PARAM(30);
  const float* dt_q_s     = PARAM(31); const float* dt_k_s    = PARAM(32);
  const float* dt_proj_w  = PARAM(33); const float* dt_proj_b = PARAM(34);
  const float* dt_mlp0_w  = PARAM(35); const float* dt_mlp0_b = PARAM(36);
  const float* dt_mlp2_w  = PARAM(37); const float* dt_mlp2_b = PARAM(38);
  const float* s1_w       = PARAM(39); const float* s1_b      = PARAM(40);
  const float* s2_w       = PARAM(41); const float* s2_b      = PARAM(42);
  const float* s_q_s      = PARAM(43); const float* s_k_s     = PARAM(44);
  const float* fin_w      = PARAM(45); const float* fin_b     = PARAM(46);
#undef PARAM

  char* wsb = (char*)d_ws;
  size_t used = 0;
  auto alloc = [&](size_t nbytes) -> void* {
    void* p = wsb + used;
    used += (nbytes + 255) & ~(size_t)255;
    return p;
  };

  // f16 transposed weights [N][K]
  f16* w_img_in  = (f16*)alloc((size_t)HID * 64 * 2);
  f16* w_txt_in  = (f16*)alloc((size_t)HID * 4096 * 2);
  f16* w_apx_in  = (f16*)alloc((size_t)APXD * 64 * 2);
  f16* w_apx_lin = (f16*)alloc((size_t)5 * APXD * APXD * 2);
  f16* w_apx_lout= (f16*)alloc((size_t)5 * APXD * APXD * 2);
  f16* w_apx_out = (f16*)alloc((size_t)HID * APXD * 2);
  f16* w_di_qkv  = (f16*)alloc((size_t)3 * HID * HID * 2);
  f16* w_di_proj = (f16*)alloc((size_t)HID * HID * 2);
  f16* w_di_mlp0 = (f16*)alloc((size_t)MLPD * HID * 2);
  f16* w_di_mlp2 = (f16*)alloc((size_t)HID * MLPD * 2);
  f16* w_dt_qkv  = (f16*)alloc((size_t)3 * HID * HID * 2);
  f16* w_dt_proj = (f16*)alloc((size_t)HID * HID * 2);
  f16* w_dt_mlp0 = (f16*)alloc((size_t)MLPD * HID * 2);
  f16* w_dt_mlp2 = (f16*)alloc((size_t)HID * MLPD * 2);
  f16* w_s1      = (f16*)alloc((size_t)(3 * HID + MLPD) * HID * 2);
  f16* w_s2      = (f16*)alloc((size_t)HID * (HID + MLPD) * 2);
  f16* w_fin     = (f16*)alloc((size_t)64 * HID * 2);

  // activations
  f16*   img16  = (f16*)alloc((size_t)LI * 64 * 2);
  f16*   txt16  = (f16*)alloc((size_t)LT * 4096 * 2);
  float* apxin  = (float*)alloc((size_t)MODL * 64 * 4);
  f16*   apxin16= (f16*)alloc((size_t)MODL * 64 * 2);
  float* apxX   = (float*)alloc((size_t)MODL * APXD * 4);
  f16*   apxH16 = (f16*)alloc((size_t)MODL * APXD * 2);
  f16*   apxT16 = (f16*)alloc((size_t)MODL * APXD * 2);
  float* modv   = (float*)alloc((size_t)MODL * HID * 4);
  float* X      = (float*)alloc((size_t)LL * HID * 4);
  f16*   A16    = (f16*)alloc((size_t)LL * HID * 2);
  float* QKVM   = (float*)alloc((size_t)LL * (3 * HID + MLPD) * 4);
  f16*   Qh     = (f16*)alloc((size_t)NH * LL * HDIM * 2);
  f16*   Kh     = (f16*)alloc((size_t)NH * LL * HDIM * 2);
  f16*   Vth    = (f16*)alloc((size_t)NH * HDIM * LL * 2);
  float* Sb     = (float*)alloc((size_t)NH * LL * LL * 4);
  f16*   Pb     = (f16*)alloc((size_t)NH * LL * LL * 2);
  f16*   Hmlp   = (f16*)alloc((size_t)LL * MLPD * 2);
  f16*   ACT16  = (f16*)alloc((size_t)LL * (HID + MLPD) * 2);

  auto wt = [&](const float* src, f16* dst, int K, int N) {
    dim3 g((N + 31) / 32, (K + 31) / 32);
    wt_kernel<<<g, 256, 0, stream>>>(src, dst, K, N);
  };
  auto gemm = [&](const f16* A, const f16* Bt, void* C,
                  const float* bias, const float* gate, const float* res,
                  int M, int N, int K, int ldc, int act, int outHalf, int doClip,
                  int batch, long sA, long sB, long sC) {
    dim3 g((N + 127) / 128, (M + 127) / 128, batch);
    gemm_f16_kernel<<<g, 256, 0, stream>>>(A, Bt, C, bias, gate, res,
                                           M, N, K, ldc, sA, sB, sC,
                                           act, outHalf, doClip);
  };
  auto normmod = [&](const float* src, f16* dst, const float* shift,
                     const float* scale, int rows, int D, int mode) {
    norm_mod_kernel<<<rows, 256, 0, stream>>>(src, dst, shift, scale, D, mode);
  };
  auto ew = [&](const float* src, long sld, f16* dst, long dld,
                int cols, long rows, int act) {
    long total = rows * (long)cols;
    ew_cvt_kernel<<<(unsigned)((total + 255) / 256), 256, 0, stream>>>(
        src, sld, dst, dld, cols, total, act);
  };
  auto modr = [&](int i) { return modv + (size_t)i * HID; };

  // ---- weight conversion (f32 [K][N] -> f16 [N][K]) ----
  wt(img_in_w, w_img_in, 64, HID);
  wt(txt_in_w, w_txt_in, 4096, HID);
  wt(apx_in_w, w_apx_in, 64, APXD);
  for (int i = 0; i < 5; ++i) {
    wt(apx_l_in_w  + (size_t)i * APXD * APXD, w_apx_lin  + (size_t)i * APXD * APXD, APXD, APXD);
    wt(apx_l_out_w + (size_t)i * APXD * APXD, w_apx_lout + (size_t)i * APXD * APXD, APXD, APXD);
  }
  wt(apx_out_w, w_apx_out, APXD, HID);
  wt(di_qkv_w,  w_di_qkv,  HID, 3 * HID);
  wt(di_proj_w, w_di_proj, HID, HID);
  wt(di_mlp0_w, w_di_mlp0, HID, MLPD);
  wt(di_mlp2_w, w_di_mlp2, MLPD, HID);
  wt(dt_qkv_w,  w_dt_qkv,  HID, 3 * HID);
  wt(dt_proj_w, w_dt_proj, HID, HID);
  wt(dt_mlp0_w, w_dt_mlp0, HID, MLPD);
  wt(dt_mlp2_w, w_dt_mlp2, MLPD, HID);
  wt(s1_w,  w_s1,  HID, 3 * HID + MLPD);
  wt(s2_w,  w_s2,  HID + MLPD, HID);
  wt(fin_w, w_fin, HID, 64);

  // ---- input projections ----
  ew(in_img, 64,   img16, 64,   64,   LI, 0);
  ew(in_txt, 4096, txt16, 4096, 4096, LT, 0);
  float* Ximg = X + (size_t)LT * HID;
  gemm(img16, w_img_in, Ximg, img_in_b, nullptr, nullptr,
       LI, HID, 64, HID, 0, 0, 0, 1, 0, 0, 0);
  gemm(txt16, w_txt_in, X, txt_in_b, nullptr, nullptr,
       LT, HID, 4096, HID, 0, 0, 0, 1, 0, 0, 0);

  // ---- approximator -> modulation vectors ----
  emb_kernel<<<(MODL * 64 + 255) / 256, 256, 0, stream>>>(in_ts, in_gd, apxin);
  ew(apxin, 64, apxin16, 64, 64, MODL, 0);
  gemm(apxin16, w_apx_in, apxX, apx_in_b, nullptr, nullptr,
       MODL, APXD, 64, APXD, 0, 0, 0, 1, 0, 0, 0);
  for (int i = 0; i < 5; ++i) {
    normmod(apxX, apxH16, nullptr, apx_norm_s + (size_t)i * APXD, MODL, APXD, 1);
    gemm(apxH16, w_apx_lin + (size_t)i * APXD * APXD, apxT16,
         apx_l_in_b + (size_t)i * APXD, nullptr, nullptr,
         MODL, APXD, APXD, APXD, 2 /*silu*/, 1, 0, 1, 0, 0, 0);
    gemm(apxT16, w_apx_lout + (size_t)i * APXD * APXD, apxX,
         apx_l_out_b + (size_t)i * APXD, nullptr, apxX,
         MODL, APXD, APXD, APXD, 0, 0, 0, 1, 0, 0, 0);
  }
  ew(apxX, APXD, apxH16, APXD, APXD, MODL, 0);
  gemm(apxH16, w_apx_out, modv, apx_out_b, nullptr, nullptr,
       MODL, HID, APXD, HID, 0, 0, 0, 1, 0, 0, 0);

  const float attn_scale = 0.08838834764831845f; // 1/sqrt(128)
  const long sQ = (long)LL * HDIM, sS = (long)LL * LL;

  // ---- dual stream ----
  f16* A16img = A16 + (size_t)LT * HID;
  normmod(Ximg, A16img, modr(3), modr(4), LI, HID, 0);
  normmod(X,    A16,    modr(9), modr(10), LT, HID, 0);
  gemm(A16img, w_di_qkv, QKVM + (size_t)LT * 3 * HID, di_qkv_b, nullptr, nullptr,
       LI, 3 * HID, HID, 3 * HID, 0, 0, 0, 1, 0, 0, 0);
  gemm(A16, w_dt_qkv, QKVM, dt_qkv_b, nullptr, nullptr,
       LT, 3 * HID, HID, 3 * HID, 0, 0, 0, 1, 0, 0, 0);
  qkv_post_kernel<<<dim3(LL, NH), 128, 0, stream>>>(
      QKVM, 3 * HID, in_img_ids, dt_q_s, dt_k_s, di_q_s, di_k_s, Qh, Kh, Vth);
  gemm(Qh, Kh, Sb, nullptr, nullptr, nullptr,
       LL, LL, HDIM, LL, 0, 0, 0, NH, sQ, sQ, sS);
  softmax_kernel<<<dim3(LL, NH), 256, 0, stream>>>(Sb, Pb, attn_scale);
  gemm(Pb, Vth, A16, nullptr, nullptr, nullptr,
       LL, HDIM, LL, HID, 0, 1 /*f16 out*/, 0, NH, sS, sQ, (long)HDIM);
  gemm(A16img, w_di_proj, Ximg, di_proj_b, modr(5), Ximg,
       LI, HID, HID, HID, 0, 0, 0, 1, 0, 0, 0);
  gemm(A16, w_dt_proj, X, dt_proj_b, modr(11), X,
       LT, HID, HID, HID, 0, 0, 0, 1, 0, 0, 0);

  normmod(Ximg, A16img, modr(6), modr(7), LI, HID, 0);
  gemm(A16img, w_di_mlp0, Hmlp, di_mlp0_b, nullptr, nullptr,
       LI, MLPD, HID, MLPD, 1 /*gelu*/, 1, 0, 1, 0, 0, 0);
  gemm(Hmlp, w_di_mlp2, Ximg, di_mlp2_b, modr(8), Ximg,
       LI, HID, MLPD, HID, 0, 0, 0, 1, 0, 0, 0);
  normmod(X, A16, modr(12), modr(13), LT, HID, 0);
  gemm(A16, w_dt_mlp0, Hmlp, dt_mlp0_b, nullptr, nullptr,
       LT, MLPD, HID, MLPD, 1, 1, 0, 1, 0, 0, 0);
  gemm(Hmlp, w_dt_mlp2, X, dt_mlp2_b, modr(14), X,
       LT, HID, MLPD, HID, 0, 0, 0, 1, 0, 0, 0);
  clip_kernel<<<((long)LT * HID + 255) / 256, 256, 0, stream>>>(X, (long)LT * HID);

  // ---- single stream ----
  normmod(X, A16, modr(0), modr(1), LL, HID, 0);
  gemm(A16, w_s1, QKVM, s1_b, nullptr, nullptr,
       LL, 3 * HID + MLPD, HID, 3 * HID + MLPD, 0, 0, 0, 1, 0, 0, 0);
  qkv_post_kernel<<<dim3(LL, NH), 128, 0, stream>>>(
      QKVM, 3 * HID + MLPD, in_img_ids, s_q_s, s_k_s, s_q_s, s_k_s, Qh, Kh, Vth);
  gemm(Qh, Kh, Sb, nullptr, nullptr, nullptr,
       LL, LL, HDIM, LL, 0, 0, 0, NH, sQ, sQ, sS);
  softmax_kernel<<<dim3(LL, NH), 256, 0, stream>>>(Sb, Pb, attn_scale);
  gemm(Pb, Vth, ACT16, nullptr, nullptr, nullptr,
       LL, HDIM, LL, HID + MLPD, 0, 1, 0, NH, sS, sQ, (long)HDIM);
  ew(QKVM + 3 * HID, 3 * HID + MLPD, ACT16 + HID, HID + MLPD, MLPD, LL, 1 /*gelu*/);
  gemm(ACT16, w_s2, X, s2_b, modr(2), X,
       LL, HID, HID + MLPD, HID, 0, 0, 1 /*clip*/, 1, 0, 0, 0);

  // ---- final projection (img rows only) ----
  normmod(Ximg, A16, modr(15), modr(16), LI, HID, 0);
  gemm(A16, w_fin, (float*)d_out, fin_b, nullptr, nullptr,
       LI, 64, HID, 64, 0, 0, 0, 1, 0, 0, 0);
}
